// MultiHeadAttention_66984309948505
// MI455X (gfx1250) — matrix operand order repaired and checked by reading, not compiled, not run on hardware
//
#include <hip/hip_runtime.h>
#include <stdint.h>

#define NUM_HEADS 16
#define N_EMBD    1024
#define HEAD_SIZE 64
#define BATCH     4
#define SEQ       2048

typedef __attribute__((ext_vector_type(16))) __bf16 v16bf;
typedef __attribute__((ext_vector_type(8)))  float  v8f;

union FragAB {
    uint4 q[2];
    unsigned short us[16];
    v16bf b;
};

__device__ __forceinline__ unsigned short f2bf(float f) {
    union { float f; unsigned u; } v; v.f = f;
    unsigned r = v.u + 0x7FFFu + ((v.u >> 16) & 1u);   // round-to-nearest-even
    return (unsigned short)(r >> 16);
}

#define WMMA_BF16(A, B, C) \
    __builtin_amdgcn_wmma_f32_16x16x32_bf16(false, (A), false, (B), (short)0, (C), false, false)

// ---------------------------------------------------------------- pack kernels
__global__ void k_cvt_bf16(const float* __restrict__ in, unsigned short* __restrict__ out, int n) {
    int i = blockIdx.x * blockDim.x + threadIdx.x;
    int stride = gridDim.x * blockDim.x;
    for (; i < n; i += stride) out[i] = f2bf(in[i]);
}

// in: [H, D, hs] fp32 -> out: [H, hs, D] bf16 (column-contiguous for B fragments)
__global__ void k_transpose_w(const float* __restrict__ in, unsigned short* __restrict__ out) {
    int i = blockIdx.x * blockDim.x + threadIdx.x;
    const int n = NUM_HEADS * N_EMBD * HEAD_SIZE;
    if (i >= n) return;
    int h   = i / (N_EMBD * HEAD_SIZE);
    int rem = i - h * (N_EMBD * HEAD_SIZE);
    int d   = rem / HEAD_SIZE;
    int s   = rem - d * HEAD_SIZE;
    out[((size_t)h * HEAD_SIZE + s) * N_EMBD + d] = f2bf(in[i]);
}

// ---------------------------------------------------------------- QKV projection
// grid (B*T/32, H), block 128 (4 waves). Wave w owns the 16-wide hs slice s0=w*16 and a
// 32-row tile: per k-step 2 A frags + 3 B frags feed 6 WMMAs (B reused across row tiles).
// q,k stored [B,H,T,hs]; v stored transposed [B,H,hs,T].
__global__ __launch_bounds__(128) void k_qkv(
    const unsigned short* __restrict__ xb,
    const unsigned short* __restrict__ wqt,
    const unsigned short* __restrict__ wkt,
    const unsigned short* __restrict__ wvt,
    unsigned short* __restrict__ qo,
    unsigned short* __restrict__ ko,
    unsigned short* __restrict__ vto) {
    const int row0 = blockIdx.x * 32;
    const int h    = blockIdx.y;
    const int wave = threadIdx.x >> 5;
    const int lane = threadIdx.x & 31;
    const int s0   = wave * 16;

    const int am  = lane & 15;           // A fragment: row within tile
    const int akb = (lane >> 4) * 8;     // A fragment: k chunk base (chunked 8+8 layout)
    const int bn  = lane & 15;           // B fragment: column
    const int bko = (lane >> 4) * 8;     // B fragment: k chunk base (chunked 8+8 layout)

    const unsigned short* aBase0 = xb  + (size_t)(row0 + am) * N_EMBD + akb;
    const unsigned short* aBase1 = xb  + (size_t)(row0 + 16 + am) * N_EMBD + akb;
    const unsigned short* bqBase = wqt + (size_t)(h * HEAD_SIZE + s0 + bn) * N_EMBD + bko;
    const unsigned short* bkBase = wkt + (size_t)(h * HEAD_SIZE + s0 + bn) * N_EMBD + bko;
    const unsigned short* bvBase = wvt + (size_t)(h * HEAD_SIZE + s0 + bn) * N_EMBD + bko;

    v8f cq[2] = {v8f{}, v8f{}};
    v8f ck[2] = {v8f{}, v8f{}};
    v8f cv[2] = {v8f{}, v8f{}};
    for (int k0 = 0; k0 < N_EMBD; k0 += 32) {
        FragAB a0, a1;
        a0.q[0] = *(const uint4*)(aBase0 + k0);  a0.q[1] = *(const uint4*)(aBase0 + k0 + 16);
        a1.q[0] = *(const uint4*)(aBase1 + k0);  a1.q[1] = *(const uint4*)(aBase1 + k0 + 16);
        FragAB bq, bk, bv;
        bq.q[0] = *(const uint4*)(bqBase + k0);  bq.q[1] = *(const uint4*)(bqBase + k0 + 16);
        bk.q[0] = *(const uint4*)(bkBase + k0);  bk.q[1] = *(const uint4*)(bkBase + k0 + 16);
        bv.q[0] = *(const uint4*)(bvBase + k0);  bv.q[1] = *(const uint4*)(bvBase + k0 + 16);
        cq[0] = WMMA_BF16(a0.b, bq.b, cq[0]);
        cq[1] = WMMA_BF16(a1.b, bq.b, cq[1]);
        ck[0] = WMMA_BF16(a0.b, bk.b, ck[0]);
        ck[1] = WMMA_BF16(a1.b, bk.b, ck[1]);
        cv[0] = WMMA_BF16(a0.b, bv.b, cv[0]);
        cv[1] = WMMA_BF16(a1.b, bv.b, cv[1]);
    }

    const int b  = row0 >> 11;           // row0 / SEQ (tiles never cross batch: 2048 % 32 == 0)
    const int bh = b * NUM_HEADS + h;
    const int mb = (lane >> 4) * 8;      // C fragment: row base for this half-wave

    for (int i = 0; i < 2; ++i) {
        const int t0 = ((row0 + i * 16) & (SEQ - 1));
        // q, k: row-major [bh, t, s]
        for (int r = 0; r < 8; ++r) {
            int t = t0 + mb + r;
            qo[((size_t)bh * SEQ + t) * HEAD_SIZE + s0 + bn] = f2bf(cq[i][r]);
            ko[((size_t)bh * SEQ + t) * HEAD_SIZE + s0 + bn] = f2bf(ck[i][r]);
        }
        // v transposed: [bh, s, t] ; lane owns one s-column, 8 consecutive t -> one 16B store
        union { uint4 v; unsigned short s[8]; } pack;
        for (int r = 0; r < 8; ++r) pack.s[r] = f2bf(cv[i][r]);
        *(uint4*)(vto + ((size_t)bh * HEAD_SIZE + s0 + bn) * SEQ + t0 + mb) = pack.v;
    }
}

// ---------------------------------------------------------------- flash attention
// grid (T/16, B*H), block 32 (1 wave per 16-query tile). Online softmax over 32-key chunks.
__global__ __launch_bounds__(32) void k_attn(
    const unsigned short* __restrict__ qb,
    const unsigned short* __restrict__ kbuf,
    const unsigned short* __restrict__ vt,
    unsigned short* __restrict__ Y) {
    const int qt  = blockIdx.x;
    const int bh  = blockIdx.y;
    const int b   = bh / NUM_HEADS;
    const int h   = bh - b * NUM_HEADS;
    const int q0  = qt * 16;
    const int lane = threadIdx.x & 31;

    __shared__ float          sS[16][36];     // raw scores (padded)
    __shared__ unsigned short sP[16][40];     // softmaxed P in bf16
    __shared__ float          sAlpha[16];
    __shared__ float          sL[16];

    const int am  = lane & 15;
    const int akb = (lane >> 4) * 8;
    const int bn  = lane & 15;
    const int bko = (lane >> 4) * 8;
    const int mb  = (lane >> 4) * 8;

    // Q A-fragments (hs = 64 -> two k=32 fragments), resident for whole tile
    const unsigned short* qrow = qb + ((size_t)bh * SEQ + q0 + am) * HEAD_SIZE + akb;
    FragAB qa0, qa1;
    qa0.q[0] = *(const uint4*)(qrow);       qa0.q[1] = *(const uint4*)(qrow + 16);
    qa1.q[0] = *(const uint4*)(qrow + 32);  qa1.q[1] = *(const uint4*)(qrow + 48);

    v8f acc[4] = {v8f{}, v8f{}, v8f{}, v8f{}};
    float m_run = -3.0e38f, l_run = 0.f;
    const float scale = 0.125f;                       // 1/sqrt(64)
    const int nchunks = (q0 + 16 + 31) >> 5;

    for (int jc = 0; jc < nchunks; ++jc) {
        const int jk = jc * 32;

        // ---- scores: two 16x16 tiles (keys jk..jk+15, jk+16..jk+31)
        v8f sc[2];
        for (int jt = 0; jt < 2; ++jt) {
            const unsigned short* krow =
                kbuf + ((size_t)bh * SEQ + jk + jt * 16 + bn) * HEAD_SIZE + bko;
            FragAB kb0, kb1;
            kb0.q[0] = *(const uint4*)(krow);       kb0.q[1] = *(const uint4*)(krow + 16);
            kb1.q[0] = *(const uint4*)(krow + 32);  kb1.q[1] = *(const uint4*)(krow + 48);
            v8f c = {};
            c = WMMA_BF16(qa0.b, kb0.b, c);
            c = WMMA_BF16(qa1.b, kb1.b, c);
            sc[jt] = c;
        }

        __syncthreads();   // previous iteration's sP/sS reads complete
        for (int r = 0; r < 8; ++r) {
            sS[mb + r][bn]      = sc[0][r] * scale;
            sS[mb + r][16 + bn] = sc[1][r] * scale;
        }
        __syncthreads();

        // ---- online softmax: 16 lanes, one query row each
        if (lane < 16) {
            const int row  = lane;
            const int qpos = q0 + row;
            float mx = m_run;
            for (int kk = 0; kk < 32; ++kk) {
                float s = (jk + kk > qpos) ? -3.0e38f : sS[row][kk];
                mx = fmaxf(mx, s);
            }
            float alpha = __expf(m_run - mx);
            float sum = 0.f;
            for (int kk = 0; kk < 32; ++kk) {
                float p = (jk + kk > qpos) ? 0.f : __expf(sS[row][kk] - mx);
                sum += p;
                sP[row][kk] = f2bf(p);
            }
            l_run = l_run * alpha + sum;
            m_run = mx;
            sAlpha[row] = alpha;
        }
        __syncthreads();

        // ---- rescale accumulators, then O += P @ V
        for (int t = 0; t < 4; ++t)
            for (int r = 0; r < 8; ++r)
                acc[t][r] *= sAlpha[mb + r];

        FragAB pa;   // C-layout -> A-layout transpose via LDS
        for (int j = 0; j < 8; ++j) {
            pa.us[j]     = sP[am][akb + j];
            pa.us[j + 8] = sP[am][akb + 16 + j];
        }
        for (int t = 0; t < 4; ++t) {
            const unsigned short* vrow =
                vt + ((size_t)bh * HEAD_SIZE + t * 16 + bn) * SEQ + jk + bko;
            FragAB vb;
            vb.q[0] = *(const uint4*)(vrow);  vb.q[1] = *(const uint4*)(vrow + 16);
            acc[t] = WMMA_BF16(pa.b, vb.b, acc[t]);
        }
    }

    if (lane < 16) sL[lane] = l_run;
    __syncthreads();

    // ---- normalize and write Y = [B, T, H*hs] bf16
    for (int t = 0; t < 4; ++t) {
        int col = h * HEAD_SIZE + t * 16 + bn;
        for (int r = 0; r < 8; ++r) {
            int trow = q0 + mb + r;
            float o = acc[t][r] / sL[mb + r];
            Y[((size_t)b * SEQ + trow) * N_EMBD + col] = f2bf(o);
        }
    }
}

// ---------------------------------------------------------------- output projection
// out = Y @ Wo^T + bo ; grid (B*T/32, D/32), 1 wave per 32x32 tile:
// per k-step 2 A frags + 2 B frags feed 4 WMMAs (2x reuse each way).
__global__ __launch_bounds__(32) void k_oproj(
    const unsigned short* __restrict__ Y,
    const unsigned short* __restrict__ wo,
    const float* __restrict__ bo,
    float* __restrict__ out) {
    const int row0 = blockIdx.x * 32;
    const int n0   = blockIdx.y * 32;
    const int lane = threadIdx.x & 31;
    const int am   = lane & 15;
    const int akb  = (lane >> 4) * 8;
    const int bn   = lane & 15;
    const int bko  = (lane >> 4) * 8;

    const unsigned short* aBase0 = Y  + (size_t)(row0 + am) * N_EMBD + akb;
    const unsigned short* aBase1 = Y  + (size_t)(row0 + 16 + am) * N_EMBD + akb;
    const unsigned short* bBase0 = wo + (size_t)(n0 + bn) * N_EMBD + bko;        // Wo rows = B cols
    const unsigned short* bBase1 = wo + (size_t)(n0 + 16 + bn) * N_EMBD + bko;

    v8f c00 = {}; v8f c01 = {}; v8f c10 = {}; v8f c11 = {};
    for (int k0 = 0; k0 < N_EMBD; k0 += 32) {
        FragAB a0, a1;
        a0.q[0] = *(const uint4*)(aBase0 + k0);  a0.q[1] = *(const uint4*)(aBase0 + k0 + 16);
        a1.q[0] = *(const uint4*)(aBase1 + k0);  a1.q[1] = *(const uint4*)(aBase1 + k0 + 16);
        FragAB b0, b1;
        b0.q[0] = *(const uint4*)(bBase0 + k0);  b0.q[1] = *(const uint4*)(bBase0 + k0 + 16);
        b1.q[0] = *(const uint4*)(bBase1 + k0);  b1.q[1] = *(const uint4*)(bBase1 + k0 + 16);
        c00 = WMMA_BF16(a0.b, b0.b, c00);
        c01 = WMMA_BF16(a0.b, b1.b, c01);
        c10 = WMMA_BF16(a1.b, b0.b, c10);
        c11 = WMMA_BF16(a1.b, b1.b, c11);
    }

    const float bias0 = bo[n0 + bn];
    const float bias1 = bo[n0 + 16 + bn];
    const int mb = (lane >> 4) * 8;
    for (int r = 0; r < 8; ++r) {
        out[(size_t)(row0 + mb + r) * N_EMBD + n0 + bn]           = c00[r] + bias0;
        out[(size_t)(row0 + mb + r) * N_EMBD + n0 + 16 + bn]      = c01[r] + bias1;
        out[(size_t)(row0 + 16 + mb + r) * N_EMBD + n0 + bn]      = c10[r] + bias0;
        out[(size_t)(row0 + 16 + mb + r) * N_EMBD + n0 + 16 + bn] = c11[r] + bias1;
    }
}

// ---------------------------------------------------------------- launch
extern "C" void kernel_launch(void* const* d_in, const int* in_sizes, int n_in,
                              void* d_out, int out_size, void* d_ws, size_t ws_size,
                              hipStream_t stream) {
    const float* x  = (const float*)d_in[0];
    const float* Wq = (const float*)d_in[1];
    const float* Wk = (const float*)d_in[2];
    const float* Wv = (const float*)d_in[3];
    const float* Wo = (const float*)d_in[4];
    const float* bo = (const float*)d_in[5];
    float* out = (float*)d_out;

    char* ws = (char*)d_ws;
    const size_t XB = (size_t)BATCH * SEQ * N_EMBD * 2;                 // 16 MB
    const size_t WT = (size_t)NUM_HEADS * HEAD_SIZE * N_EMBD * 2;       //  2 MB
    const size_t QS = (size_t)BATCH * NUM_HEADS * SEQ * HEAD_SIZE * 2;  // 16 MB

    unsigned short* xb   = (unsigned short*)(ws);
    unsigned short* wqt  = (unsigned short*)(ws + XB);
    unsigned short* wkt  = (unsigned short*)(ws + XB + WT);
    unsigned short* wvt  = (unsigned short*)(ws + XB + 2 * WT);
    unsigned short* wob  = (unsigned short*)(ws + XB + 3 * WT);
    unsigned short* qb   = (unsigned short*)(ws + XB + 4 * WT);
    unsigned short* kbuf = (unsigned short*)(ws + XB + 4 * WT + QS);
    unsigned short* vtb  = (unsigned short*)(ws + XB + 4 * WT + 2 * QS);
    unsigned short* Y    = xb;   // alias: xb fully consumed by k_qkv before k_attn writes Y

    const int nx = BATCH * SEQ * N_EMBD;
    k_cvt_bf16<<<512, 256, 0, stream>>>(x, xb, nx);
    const int nw = NUM_HEADS * N_EMBD * HEAD_SIZE;
    k_transpose_w<<<(nw + 255) / 256, 256, 0, stream>>>(Wq, wqt);
    k_transpose_w<<<(nw + 255) / 256, 256, 0, stream>>>(Wk, wkt);
    k_transpose_w<<<(nw + 255) / 256, 256, 0, stream>>>(Wv, wvt);
    k_cvt_bf16<<<(N_EMBD * N_EMBD + 255) / 256, 256, 0, stream>>>(Wo, wob, N_EMBD * N_EMBD);

    k_qkv<<<dim3((BATCH * SEQ) / 32, NUM_HEADS), 128, 0, stream>>>(xb, wqt, wkt, wvt, qb, kbuf, vtb);
    k_attn<<<dim3(SEQ / 16, BATCH * NUM_HEADS), 32, 0, stream>>>(qb, kbuf, vtb, Y);
    k_oproj<<<dim3((BATCH * SEQ) / 32, N_EMBD / 32), 32, 0, stream>>>(Y, wob, bo, out);
}
